// VAE_5471788335518
// MI455X (gfx1250) — compile-verified, matrix operand order checked by reading, not run on hardware
//
#include <hip/hip_runtime.h>
#include <hip/hip_bf16.h>
#include <stdint.h>

// ---------------------------------------------------------------------------
// Problem constants
// ---------------------------------------------------------------------------
#define VOC 32000
#define EMB 512
#define HID 512
#define BAT 64
#define LPR 64
#define LHY 64
#define G4  2048   // 4*HID
#define NT_LOGIT (VOC / 64)   // 500 blocks of 64 columns

typedef __attribute__((ext_vector_type(16))) __bf16 v16bf;
typedef __attribute__((ext_vector_type(8)))  float  v8f;

// ---------------------------------------------------------------------------
// WMMA fragment helpers (16x16x32 bf16, f32 accumulate)
//
// 16-bit A-matrix 16x32 layout (ISA 7.12.2): lane L in [0,15] holds row L,
// K = [0..7] and [16..23]; lane L+16 holds row L, K = [8..15] and [24..31].
// For B = W^T with W row-major (N x K), B[k][n] = W[n][k], so the B fragment
// is the *same* contiguous-per-row pattern on W: two 16B loads per lane
// (global_load_b128 from global, ds_load_b128 from LDS).
// ---------------------------------------------------------------------------
union FragU { v16bf v; uint4 q[2]; };

__device__ __forceinline__ v16bf load_frag(const __bf16* base, int ld, int lane) {
    const int r  = lane & 15;
    const int ko = (lane >> 4) << 3;          // 0 or 8
    const __bf16* p = base + (size_t)r * ld + ko;
    FragU u;
    u.q[0] = *(const uint4*)(p);              // K = ko .. ko+7
    u.q[1] = *(const uint4*)(p + 16);         // K = ko+16 .. ko+23
    return u.v;
}

__device__ __forceinline__ v8f wmma_bf16(v16bf a, v16bf b, v8f c) {
    return __builtin_amdgcn_wmma_f32_16x16x32_bf16(false, a, false, b, (short)0, c, false, false);
}

__device__ __forceinline__ float sigmoidf_(float x) { return 1.0f / (1.0f + __expf(-x)); }

// ---------------------------------------------------------------------------
// Fully-unrolled K-loop, B from GLOBAL (one-shot kernels).
// ---------------------------------------------------------------------------
template <int Q>
__device__ __forceinline__ void gemm_pipe(const __bf16* __restrict__ Ab,
                                          const __bf16* const (&Bb)[Q],
                                          int lane, v8f* acc) {
#pragma unroll
    for (int k = 0; k < 512; k += 32) {
        v16bf a = load_frag(Ab + k, 512, lane);
        v16bf b[Q];
#pragma unroll
        for (int q = 0; q < Q; ++q) b[q] = load_frag(Bb[q] + k, 512, lane);
#pragma unroll
        for (int q = 0; q < Q; ++q) acc[q] = wmma_bf16(a, b[q], acc[q]);
    }
}

// ---------------------------------------------------------------------------
// Fully-unrolled K-loop, B from LDS (per-step kernels). The B panel is
// staged once per block (shared by all 4 waves -> 4x less L2 traffic);
// inner-loop B reads are short-latency ds_load_b128.
// Bs holds Q tiles of 16 rows x 512 cols, row-major.
// ---------------------------------------------------------------------------
template <int Q>
__device__ __forceinline__ void gemm_lds(const __bf16* __restrict__ Ab,
                                         const __bf16* Bs, int lane, v8f* acc) {
#pragma unroll
    for (int k = 0; k < 512; k += 32) {
        v16bf a = load_frag(Ab + k, 512, lane);
        v16bf b[Q];
#pragma unroll
        for (int q = 0; q < Q; ++q) b[q] = load_frag(Bs + q * 16 * 512 + k, 512, lane);
#pragma unroll
        for (int q = 0; q < Q; ++q) acc[q] = wmma_bf16(a, b[q], acc[q]);
    }
}

// ---------------------------------------------------------------------------
// fp32 -> bf16 conversion (grid-stride)
// ---------------------------------------------------------------------------
__global__ void k_cvt_bf16(const float* __restrict__ s, __bf16* __restrict__ d, int n) {
    int i = blockIdx.x * blockDim.x + threadIdx.x;
    int stride = gridDim.x * blockDim.x;
    for (; i < n; i += stride) d[i] = (__bf16)s[i];
}

// ---------------------------------------------------------------------------
// Embedding gather, step-major rows: dst[(t*64+b)][e] = embed[tok[b][t]][e]
// ---------------------------------------------------------------------------
__global__ void k_gather(const float* __restrict__ embed, const int* __restrict__ toks,
                         __bf16* __restrict__ dst, int ldtok) {
    const int row = blockIdx.x;               // t*64 + b
    const int t = row >> 6, b = row & 63;
    const int tok = toks[b * ldtok + t];
    const float* src = embed + (size_t)tok * EMB;
    __bf16* d = dst + (size_t)row * EMB;
    for (int i = threadIdx.x; i < EMB; i += blockDim.x) d[i] = (__bf16)src[i];
}

// ---------------------------------------------------------------------------
// Hoisted GEMM: C[M x N] = A[M x 512] * B[N x 512]^T + b1 + b2 (fp32 out)
// block = 128 threads (4 waves); grid = (N/64, M/64). Each wave owns a
// 16x64 output strip: per K-iter, 1 A-frag + 4 B-frags -> 4 WMMAs.
// ---------------------------------------------------------------------------
__global__ void k_gemm_bias(const __bf16* __restrict__ A, const __bf16* __restrict__ B,
                            const float* __restrict__ b1, const float* __restrict__ b2,
                            float* __restrict__ C, int N) {
    const int lane = threadIdx.x & 31;
    const int wave = threadIdx.x >> 5;
    const int mbase = blockIdx.y * 64 + wave * 16;
    const int j0 = blockIdx.x * 64;
    const int n = lane & 15, hi = lane >> 4;
    v8f acc[4];
#pragma unroll
    for (int q = 0; q < 4; ++q) {
        const float bias = b1[j0 + 16 * q + n] + b2[j0 + 16 * q + n];
#pragma unroll
        for (int v = 0; v < 8; ++v) acc[q][v] = bias;
    }
    const __bf16* Ab = A + (size_t)mbase * 512;
    const __bf16* Bb[4] = {B + (size_t)(j0 + 0) * 512,  B + (size_t)(j0 + 16) * 512,
                           B + (size_t)(j0 + 32) * 512, B + (size_t)(j0 + 48) * 512};
    gemm_pipe<4>(Ab, Bb, lane, acc);
#pragma unroll
    for (int q = 0; q < 4; ++q) {
        float* Cp = C + (size_t)mbase * N + j0 + 16 * q + n;
#pragma unroll
        for (int v = 0; v < 8; ++v) Cp[(size_t)(v + 8 * hi) * N] = acc[q][v];
    }
}

// ---------------------------------------------------------------------------
// Fused LSTM step: g = xW_t + h*Whh^T ; cell update ; h_out = bf16(h)
// grid = 32 blocks (col tiles of HID) x 128 threads (4 waves = 4 row tiles).
// The 4 gate weight tiles (64 rows of Whh) are staged in LDS once per
// block -- all 4 waves share them. Accumulators init from hoisted x-part.
// ---------------------------------------------------------------------------
__global__ void k_lstm_step(const __bf16* __restrict__ h_in, const float* __restrict__ xW,
                            const __bf16* __restrict__ Whh, float* __restrict__ c,
                            __bf16* __restrict__ h_out) {
    __shared__ __bf16 Bs[4 * 16 * 512];       // 64 KB
    const int lane = threadIdx.x & 31;
    const int wave = threadIdx.x >> 5;
    const int m0 = wave * 16;
    const int j0 = blockIdx.x * 16;
    const int n = lane & 15, hi = lane >> 4;

    // Stage Whh rows {q*512 + j0 .. +15} for q=0..3 into LDS (16B chunks).
    {
        uint4* dst = (uint4*)Bs;
        for (int i = threadIdx.x; i < 4096; i += 128) {
            const int q = i >> 10;            // 1024 uint4 per 16-row tile
            const int r = i & 1023;
            dst[i] = ((const uint4*)(Whh + (size_t)(q * 512 + j0) * 512))[r];
        }
    }

    v8f acc[4];
#pragma unroll
    for (int q = 0; q < 4; ++q) {
        const float* xp = xW + (size_t)m0 * G4 + q * 512 + j0 + n;
#pragma unroll
        for (int v = 0; v < 8; ++v) acc[q][v] = xp[(size_t)(v + 8 * hi) * G4];
    }
    __syncthreads();

    const __bf16* Ab = h_in + (size_t)m0 * 512;
    gemm_lds<4>(Ab, Bs, lane, acc);

#pragma unroll
    for (int v = 0; v < 8; ++v) {
        const int m = m0 + v + 8 * hi;
        const size_t idx = (size_t)m * HID + j0 + n;
        const float gi = acc[0][v], gf = acc[1][v], gg = acc[2][v], go = acc[3][v];
        const float cn = sigmoidf_(gf) * c[idx] + sigmoidf_(gi) * tanhf(gg);
        c[idx] = cn;
        h_out[idx] = (__bf16)(sigmoidf_(go) * tanhf(cn));
    }
}

// ---------------------------------------------------------------------------
// mu / logvar / z: two WMMA accumulators per wave, then reparameterization.
// grid = 32 x 128 as above. Writes mu, log_var into d_out; z as fp32 cell
// init and bf16 hidden init for the decoder.
// ---------------------------------------------------------------------------
__global__ void k_mu_z(const __bf16* __restrict__ h, const __bf16* __restrict__ muW,
                       const __bf16* __restrict__ lvW, const float* __restrict__ mub,
                       const float* __restrict__ lvb, const float* __restrict__ eps,
                       float* __restrict__ mu_out, float* __restrict__ lv_out,
                       float* __restrict__ c_out, __bf16* __restrict__ z_out) {
    const int lane = threadIdx.x & 31;
    const int wave = threadIdx.x >> 5;
    const int m0 = wave * 16;
    const int j0 = blockIdx.x * 16;
    const int n = lane & 15, hi = lane >> 4;
    v8f acc[2];
    const float bm = mub[j0 + n], bl = lvb[j0 + n];
#pragma unroll
    for (int v = 0; v < 8; ++v) { acc[0][v] = bm; acc[1][v] = bl; }
    const __bf16* Ab = h + (size_t)m0 * 512;
    const __bf16* Bb[2] = {muW + (size_t)j0 * 512, lvW + (size_t)j0 * 512};
    gemm_pipe<2>(Ab, Bb, lane, acc);
#pragma unroll
    for (int v = 0; v < 8; ++v) {
        const int m = m0 + v + 8 * hi;
        const size_t idx = (size_t)m * HID + j0 + n;
        const float mu = acc[0][v], lv = acc[1][v];
        mu_out[idx] = mu;
        lv_out[idx] = lv;
        const float z = mu + eps[idx] * __expf(0.5f * lv);
        c_out[idx] = z;
        z_out[idx] = (__bf16)z;
    }
}

// ---------------------------------------------------------------------------
// Vocab projection + per-block argmax partials.
// grid = 500 blocks (64-column vocab strips) x 128 threads (4 row waves).
// The 64-row out_W panel (64 KB) is staged in LDS once per block and shared
// by all 4 waves (4x less L2 traffic); inner loop is ds_load + WMMA.
// Logits stored straight into raw_outputs[:, t1, :]; argmax over the 64
// columns: combine 4 per-tile keys in-register, then half-wave shuffles.
// ---------------------------------------------------------------------------
__global__ void k_logits(const __bf16* __restrict__ h, const __bf16* __restrict__ W,
                         const float* __restrict__ bias, float* __restrict__ raw,
                         int t1, unsigned long long* __restrict__ partials) {
    __shared__ __bf16 Bs[4 * 16 * 512];       // 64 KB
    const int lane = threadIdx.x & 31;
    const int wave = threadIdx.x >> 5;
    const int m0 = wave * 16;
    const int j0 = blockIdx.x * 64;
    const int n = lane & 15, hi = lane >> 4;

    // Stage the 64 contiguous out_W rows for this block into LDS.
    {
        const uint4* src = (const uint4*)(W + (size_t)j0 * 512);
        uint4* dst = (uint4*)Bs;
        for (int i = threadIdx.x; i < 4096; i += 128) dst[i] = src[i];
    }

    v8f acc[4];
#pragma unroll
    for (int q = 0; q < 4; ++q) {
        const float bs = bias[j0 + 16 * q + n];
#pragma unroll
        for (int v = 0; v < 8; ++v) acc[q][v] = bs;
    }
    __syncthreads();

    const __bf16* Ab = h + (size_t)m0 * 512;
    gemm_lds<4>(Ab, Bs, lane, acc);

    // Store logits
#pragma unroll
    for (int q = 0; q < 4; ++q) {
#pragma unroll
        for (int v = 0; v < 8; ++v) {
            const int m = m0 + v + 8 * hi;
            raw[((size_t)m * LHY + t1) * VOC + j0 + 16 * q + n] = acc[q][v];
        }
    }
    // Argmax over this block's 64 columns; ties -> lowest index.
#pragma unroll
    for (int v = 0; v < 8; ++v) {
        unsigned long long key = 0ull;
#pragma unroll
        for (int q = 0; q < 4; ++q) {
            unsigned ub = __float_as_uint(acc[q][v]);
            ub = (ub & 0x80000000u) ? ~ub : (ub | 0x80000000u);  // order-preserving map
            unsigned long long kq = ((unsigned long long)ub << 32) |
                                    (unsigned)(0xFFFFFFFFu - (unsigned)(j0 + 16 * q + n));
            if (kq > key) key = kq;
        }
#pragma unroll
        for (int s = 1; s < 16; s <<= 1) {
            unsigned long long o = __shfl_xor(key, s, 32);
            if (o > key) key = o;
        }
        if (n == 0) {
            const int m = m0 + v + 8 * hi;
            partials[(size_t)blockIdx.x * BAT + m] = key;
        }
    }
}

// One block per batch row: reduce 500 partials, emit predicted token.
__global__ void k_argmax_reduce(const unsigned long long* __restrict__ partials,
                                float* __restrict__ ow, int t1) {
    const int m = blockIdx.x;
    unsigned long long best = 0ull;
    for (int i = threadIdx.x; i < NT_LOGIT; i += blockDim.x) {
        unsigned long long k = partials[(size_t)i * BAT + m];
        if (k > best) best = k;
    }
    __shared__ unsigned long long red[256];
    red[threadIdx.x] = best;
    __syncthreads();
    for (int s = blockDim.x >> 1; s > 0; s >>= 1) {
        if (threadIdx.x < s && red[threadIdx.x + s] > red[threadIdx.x])
            red[threadIdx.x] = red[threadIdx.x + s];
        __syncthreads();
    }
    if (threadIdx.x == 0) {
        unsigned idx = 0xFFFFFFFFu - (unsigned)(red[0] & 0xFFFFFFFFull);
        ow[(size_t)m * LHY + t1] = (float)idx;
    }
}

// ---------------------------------------------------------------------------
// Init: zero h0/c0; zero raw_outputs[:,0,:]; seed outputted_words[:,0].
// ---------------------------------------------------------------------------
__global__ void k_init_state(float* __restrict__ c, __bf16* __restrict__ h0) {
    const int i = blockIdx.x * blockDim.x + threadIdx.x;
    if (i < BAT * HID) { c[i] = 0.0f; h0[i] = (__bf16)0.0f; }
}

__global__ void k_init_out(float* __restrict__ raw, float* __restrict__ ow,
                           const int* __restrict__ hyp) {
    int i = blockIdx.x * blockDim.x + threadIdx.x;
    const int stride = gridDim.x * blockDim.x;
    for (; i < BAT * VOC; i += stride) {
        const int b = i / VOC, v = i - b * VOC;
        raw[(size_t)b * LHY * VOC + v] = 0.0f;       // raw_outputs[b][0][v] = 0
    }
    if (blockIdx.x == 0 && threadIdx.x < BAT)
        ow[(size_t)threadIdx.x * LHY] = (float)hyp[threadIdx.x * LHY];
}

// ---------------------------------------------------------------------------
// Host orchestration
// ---------------------------------------------------------------------------
extern "C" void kernel_launch(void* const* d_in, const int* in_sizes, int n_in,
                              void* d_out, int out_size, void* d_ws, size_t ws_size,
                              hipStream_t stream) {
    const int*   premises   = (const int*)d_in[0];
    const int*   hypothesis = (const int*)d_in[1];
    const float* embed      = (const float*)d_in[3];
    const float* enc_Wih    = (const float*)d_in[4];
    const float* enc_Whh    = (const float*)d_in[5];
    const float* enc_bih    = (const float*)d_in[6];
    const float* enc_bhh    = (const float*)d_in[7];
    const float* mu_W       = (const float*)d_in[8];
    const float* mu_b       = (const float*)d_in[9];
    const float* lv_W       = (const float*)d_in[10];
    const float* lv_b       = (const float*)d_in[11];
    const float* dec_Wih    = (const float*)d_in[12];
    const float* dec_Whh    = (const float*)d_in[13];
    const float* dec_bih    = (const float*)d_in[14];
    const float* dec_bhh    = (const float*)d_in[15];
    const float* out_W      = (const float*)d_in[16];
    const float* out_b      = (const float*)d_in[17];
    const float* eps        = (const float*)d_in[18];

    float* ow  = (float*)d_out;                               // [64][64]
    float* raw = ow + (size_t)BAT * LHY;                      // [64][64][32000]
    float* muo = raw + (size_t)BAT * LHY * VOC;               // [64][512]
    float* lvo = muo + (size_t)BAT * HID;                     // [64][512]

    // Workspace carve-up (all 256B aligned)
    char* w = (char*)d_ws;
    auto alloc = [&](size_t bytes) -> void* {
        void* p = (void*)w;
        w += (bytes + 255) & ~(size_t)255;
        return p;
    };
    __bf16* outW16   = (__bf16*)alloc((size_t)VOC * HID * 2);
    __bf16* encWih16 = (__bf16*)alloc((size_t)G4 * EMB * 2);
    __bf16* encWhh16 = (__bf16*)alloc((size_t)G4 * HID * 2);
    __bf16* decWih16 = (__bf16*)alloc((size_t)G4 * EMB * 2);
    __bf16* decWhh16 = (__bf16*)alloc((size_t)G4 * HID * 2);
    __bf16* muW16    = (__bf16*)alloc((size_t)HID * HID * 2);
    __bf16* lvW16    = (__bf16*)alloc((size_t)HID * HID * 2);
    __bf16* prem16   = (__bf16*)alloc((size_t)LPR * BAT * EMB * 2);
    __bf16* hyp16    = (__bf16*)alloc((size_t)(LHY - 1) * BAT * EMB * 2);
    float*  enc_xW   = (float*)alloc((size_t)LPR * BAT * G4 * 4);
    float*  dec_xW   = (float*)alloc((size_t)(LHY - 1) * BAT * G4 * 4);
    __bf16* hbuf     = (__bf16*)alloc(2 * (size_t)BAT * HID * 2);
    float*  cbuf     = (float*)alloc((size_t)BAT * HID * 4);
    unsigned long long* partials =
        (unsigned long long*)alloc((size_t)NT_LOGIT * BAT * 8);
    const size_t HS = (size_t)BAT * HID;   // elements per h buffer

    // --- weight conversions to bf16 ---
    k_cvt_bf16<<<2048, 256, 0, stream>>>(out_W, outW16, VOC * HID);
    k_cvt_bf16<<<512, 256, 0, stream>>>(enc_Wih, encWih16, G4 * EMB);
    k_cvt_bf16<<<512, 256, 0, stream>>>(enc_Whh, encWhh16, G4 * HID);
    k_cvt_bf16<<<512, 256, 0, stream>>>(dec_Wih, decWih16, G4 * EMB);
    k_cvt_bf16<<<512, 256, 0, stream>>>(dec_Whh, decWhh16, G4 * HID);
    k_cvt_bf16<<<256, 256, 0, stream>>>(mu_W, muW16, HID * HID);
    k_cvt_bf16<<<256, 256, 0, stream>>>(lv_W, lvW16, HID * HID);

    // --- embedding gathers (step-major rows: t*64+b) ---
    k_gather<<<LPR * BAT, 128, 0, stream>>>(embed, premises, prem16, LPR);
    k_gather<<<(LHY - 1) * BAT, 128, 0, stream>>>(embed, hypothesis, hyp16, LHY);

    // --- hoisted input-to-gates GEMMs (bias folded in) ---
    k_gemm_bias<<<dim3(G4 / 64, LPR), 128, 0, stream>>>(prem16, encWih16, enc_bih,
                                                        enc_bhh, enc_xW, G4);
    k_gemm_bias<<<dim3(G4 / 64, LHY - 1), 128, 0, stream>>>(hyp16, decWih16, dec_bih,
                                                            dec_bhh, dec_xW, G4);

    // --- state + output init ---
    k_init_state<<<(BAT * HID + 255) / 256, 256, 0, stream>>>(cbuf, hbuf);
    k_init_out<<<2048, 256, 0, stream>>>(raw, ow, hypothesis);

    // --- encoder recurrence: read hbuf[t&1], write hbuf[(t+1)&1] ---
    for (int t = 0; t < LPR; ++t) {
        k_lstm_step<<<HID / 16, 128, 0, stream>>>(
            hbuf + (size_t)(t & 1) * HS, enc_xW + (size_t)t * BAT * G4,
            encWhh16, cbuf, hbuf + (size_t)((t + 1) & 1) * HS);
    }
    // h_enc ends in hbuf[0]; z -> hbuf[1] + cbuf
    k_mu_z<<<HID / 16, 128, 0, stream>>>(hbuf, muW16, lvW16, mu_b, lv_b, eps,
                                         muo, lvo, cbuf, hbuf + HS);

    // --- decoder recurrence: read hbuf[(t+1)&1], write hbuf[t&1] ---
    for (int t = 0; t < LHY - 1; ++t) {
        __bf16* h_in  = hbuf + (size_t)((t + 1) & 1) * HS;
        __bf16* h_out = hbuf + (size_t)(t & 1) * HS;
        k_lstm_step<<<HID / 16, 128, 0, stream>>>(
            h_in, dec_xW + (size_t)t * BAT * G4, decWhh16, cbuf, h_out);
        k_logits<<<NT_LOGIT, 128, 0, stream>>>(h_out, outW16, out_b, raw, t + 1,
                                               partials);
        k_argmax_reduce<<<BAT, 256, 0, stream>>>(partials, ow, t + 1);
    }

    (void)in_sizes; (void)n_in; (void)out_size; (void)ws_size;
}
